// EDEE_40329742910074
// MI455X (gfx1250) — compile-verified, matrix operand order checked by reading, not run on hardware
//
#include <hip/hip_runtime.h>
#include <hip/hip_bf16.h>

typedef __attribute__((ext_vector_type(16))) _Float16 v16h;
typedef __attribute__((ext_vector_type(8)))  _Float16 h8;
typedef __attribute__((ext_vector_type(8)))  float    v8f;
typedef __attribute__((ext_vector_type(4)))  float    f4;
typedef __attribute__((ext_vector_type(4)))  unsigned ui4;
typedef __attribute__((ext_vector_type(8)))  int      i8v;
typedef __attribute__((ext_vector_type(4)))  int      i4v;

#define NN   384
#define INW  350
#define HH   256
#define G4   1024
#define FF   256
#define RR   16
#define PADK 264   // 256 + 8 halves pad: lane stride 132 dwords -> conflict-free b128

#if defined(__has_builtin)
#  if __has_builtin(__builtin_amdgcn_tensor_load_to_lds) && __has_builtin(__builtin_amdgcn_s_wait_tensorcnt)
#    define USE_TDM 1
#  endif
#endif

// ---------------------------------------------------------------- helpers
__device__ __forceinline__ float lrelu(float x) { return x > 0.f ? x : 0.01f * x; }
__device__ __forceinline__ float sigm(float x)  { return 1.f / (1.f + __expf(-x)); }
__device__ __forceinline__ float tanh_fast(float x) {
    float e = __expf(-2.f * x);
    return (1.f - e) / (1.f + e);
}

// Build one 16x32 f16 WMMA operand fragment from an LDS row (ISA 7.12.2 layout):
// lanes<16 -> K = ks*32 + {0..7, 16..23}; lanes>=16 -> K = ks*32 + {8..15, 24..31}
__device__ __forceinline__ v16h load_frag(const _Float16* rowbase, int ks, int khalf) {
    h8 lo = *(const h8*)(rowbase + ks * 32 + khalf * 8);
    h8 hi = *(const h8*)(rowbase + ks * 32 + 16 + khalf * 8);
    v16h r;
#pragma unroll
    for (int u = 0; u < 8; ++u) { r[u] = lo[u]; r[8 + u] = hi[u]; }
    return r;
}

#ifdef USE_TDM
// TDM: DMA a 256x256 f16 row-major tensor from global into LDS, inserting a
// 16 B pad after every 512 B row so the LDS image has a 264-half pitch
// (conflict-free ds_load_b128 fragments). D# layout per ISA ch.8:
//   group0: count=1 | lds_addr | global_addr[56:0] | type=2
//   group1: data_size=2B, pad_enable, pad_interval=6 (128 dwords), pad_amount=3 (4 dwords),
//           tensor_dim0/1=256, tile_dim0/1=256, dim0_stride=256, dim1_stride=65536
__device__ __forceinline__ void tdm_load_w2(const _Float16* gsrc, unsigned lds_off) {
    unsigned long long ga = (unsigned long long)(size_t)gsrc;
    ui4 g0;
    g0[0] = 1u;                                                  // count=1 (valid user D#)
    g0[1] = lds_off;                                             // lds_addr (bytes)
    g0[2] = (unsigned)(ga & 0xFFFFFFFFu);                        // global_addr[31:0]
    g0[3] = (unsigned)((ga >> 32) & 0x1FFFFFFu) | (2u << 30);    // global_addr[56:32] | type=2
    i8v g1;
    g1[0] = (int)((1u << 16) | (1u << 20) | (6u << 22) | (3u << 25)); // 2B elems, pad 16B/512B
    g1[1] = (int)(256u << 16);   // tensor_dim0[15:0]=256 (bits 63:48)
    g1[2] = (int)(256u << 16);   // tensor_dim0[31:16]=0, tensor_dim1[15:0]=256
    g1[3] = (int)(256u << 16);   // tensor_dim1[31:16]=0, tile_dim0=256
    g1[4] = 256;                 // tile_dim1=256, tile_dim2=0 (2D)
    g1[5] = 256;                 // tensor_dim0_stride[31:0]=256
    g1[6] = 0;                   // stride0[47:32]=0, dim1_stride[15:0]=0
    g1[7] = 1;                   // dim1_stride[47:16]=1  (65536 elements)
    i4v gz = {0, 0, 0, 0};
#if __clang_major__ >= 23
    i8v gz8 = {0, 0, 0, 0, 0, 0, 0, 0};
    __builtin_amdgcn_tensor_load_to_lds(g0, g1, gz, gz, gz8, 0);
#else
    __builtin_amdgcn_tensor_load_to_lds(g0, g1, gz, gz, 0);
#endif
}
#endif

// ---------------------------------------------------------------- kernels
__global__ void k_embed(const int* __restrict__ wid, const int* __restrict__ tyid,
                        const float* __restrict__ emb, const float* __restrict__ wty,
                        float* __restrict__ x) {
    int n = blockIdx.x;
    int w = wid[n], ty = tyid[n];
    for (int c = threadIdx.x; c < INW; c += blockDim.x)
        x[n * INW + c] = (c < 300) ? emb[w * 300 + c] : wty[ty * 50 + (c - 300)];
}

__global__ void k_tohalf(const float* __restrict__ s, _Float16* __restrict__ d, int n) {
    int i = blockIdx.x * blockDim.x + threadIdx.x;
    if (i < n) d[i] = (_Float16)s[i];
}

// gx[n][j] = (bih+bhh)[j] + dot(x[n,:350], Wih[j,:350])
__global__ __launch_bounds__(1024) void k_gx(const float* __restrict__ x,
                                             const float* __restrict__ Wih,
                                             const float* __restrict__ bih,
                                             const float* __restrict__ bhh,
                                             float* __restrict__ gx) {
    __shared__ __align__(16) float xr[352];
    int n = blockIdx.x;
    if (threadIdx.x < INW) xr[threadIdx.x] = x[n * INW + threadIdx.x];
    __syncthreads();
    int j = threadIdx.x;
    const float* w = Wih + j * INW;
    float acc = bih[j] + bhh[j];
#pragma unroll 2
    for (int k = 0; k < INW; k += 2) acc += w[k] * xr[k] + w[k + 1] * xr[k + 1];
    gx[n * G4 + j] = acc;
}

// One workgroup per direction; 384 sequential steps, h/c/gates in LDS, Whh f16 from L2.
__global__ __launch_bounds__(1024) void k_lstm(const float* __restrict__ gxf,
                                               const float* __restrict__ gxb,
                                               const _Float16* __restrict__ whhh,
                                               float* __restrict__ hcat) {
    int dir = blockIdx.x;
    int tid = threadIdx.x;
    const float* gx = dir ? gxb : gxf;
    const _Float16* wr = whhh + (size_t)dir * G4 * HH + (size_t)tid * HH;
    __shared__ float hbuf[HH], cbuf[HH], act[G4];
    if (tid < HH) { hbuf[tid] = 0.f; cbuf[tid] = 0.f; }
    __syncthreads();
    for (int t = 0; t < NN; ++t) {
        int n = dir ? (NN - 1 - t) : t;
        float acc = gx[n * G4 + tid];
#pragma unroll 8
        for (int k = 0; k < HH; k += 8) {
            h8 w = *(const h8*)(wr + k);
#pragma unroll
            for (int u = 0; u < 8; ++u) acc += (float)w[u] * hbuf[k + u];
        }
        act[tid] = ((tid >> 8) == 2) ? tanh_fast(acc) : sigm(acc);   // gate order i,f,g,o
        __syncthreads();
        if (tid < HH) {
            float c = act[HH + tid] * cbuf[tid] + act[tid] * act[2 * HH + tid];
            float h = act[3 * HH + tid] * tanh_fast(c);
            cbuf[tid] = c; hbuf[tid] = h;
            hcat[n * (2 * HH) + dir * HH + tid] = h;
        }
        __syncthreads();
    }
}

// A = h@W1[:, :512].T ; B = h@W1[:, 512:].T
__global__ __launch_bounds__(512) void k_ab(const float* __restrict__ hcat,
                                            const float* __restrict__ W1,
                                            float* __restrict__ Ao, float* __restrict__ Bo) {
    __shared__ __align__(16) float hr[512];
    int n = blockIdx.x;
    hr[threadIdx.x] = hcat[n * 512 + threadIdx.x];
    __syncthreads();
    int f = threadIdx.x & 255, half = threadIdx.x >> 8;
    const float* w = W1 + f * 1024 + half * 512;
    float acc = 0.f;
#pragma unroll 4
    for (int k = 0; k < 512; ++k) acc += w[k] * hr[k];
    (half ? Bo : Ao)[n * FF + f] = acc;
}

// Fused all-pairs: per 16-row tile (i fixed, j0..j0+15):
//   z = leaky(A[i]+B[j]+b1) -> f16 LDS
//   y = leaky(z @ W2^T + b2) via WMMA f16 -> f16 LDS
//   out = y @ Wf^T + bf via WMMA f16 -> global f32
// W2 (135 KB padded image) staged once per workgroup via the Tensor Data Mover.
__global__ __launch_bounds__(128) void k_pairs(const float* __restrict__ Ao,
                                               const float* __restrict__ Bo,
                                               const float* __restrict__ b1,
                                               const _Float16* __restrict__ w2h,
                                               const float* __restrict__ b2,
                                               const _Float16* __restrict__ wfh,
                                               const float* __restrict__ bf,
                                               float* __restrict__ out) {
    __shared__ __align__(16) _Float16 w2s[FF * PADK];      // 135168 B
    __shared__ __align__(16) _Float16 wfs[RR * PADK];      //   8448 B
    __shared__ __align__(16) _Float16 zb[4][16 * PADK];    //  33792 B
    __shared__ __align__(16) _Float16 yb[4][16 * PADK];    //  33792 B
    __shared__ __align__(16) float b1s[FF], b2s[FF], bfs[RR];

    int tid = threadIdx.x;
    int wave = tid >> 5, lane = tid & 31;
    int t  = blockIdx.x * 4 + wave;
    int i  = t / 24;
    int j0 = (t % 24) * 16;

    // warm L2 for this tile's operands while staging runs (global_prefetch_b8)
    __builtin_prefetch(&Ao[i * FF], 0, 0);
    __builtin_prefetch(&Bo[(j0 + (lane & 15)) * FF], 0, 0);

    // small tables: manual cooperative copy
    for (int idx = tid; idx < RR * 32; idx += 128) {
        int r = idx >> 5, c = (idx & 31) * 8;
        *(h8*)&wfs[r * PADK + c] = *(const h8*)&wfh[r * FF + c];
    }
    for (int idx = tid; idx < FF; idx += 128) { b1s[idx] = b1[idx]; b2s[idx] = b2[idx]; }
    if (tid < RR) bfs[tid] = bf[tid];

#ifdef USE_TDM
    // W2: one TDM descriptor does global->LDS with the 16B/row pad injected in HW.
    if (wave == 0) {
        tdm_load_w2(w2h, (unsigned)(size_t)(void*)w2s);
        __builtin_amdgcn_s_wait_tensorcnt(0);
    }
#else
    for (int idx = tid; idx < FF * 32; idx += 128) {
        int r = idx >> 5, c = (idx & 31) * 8;
        *(h8*)&w2s[r * PADK + c] = *(const h8*)&w2h[r * FF + c];
    }
#endif
    __syncthreads();

    _Float16* z = zb[wave];
    _Float16* y = yb[wave];

    // ---- build z tile: each lane owns 8 contiguous columns for all 16 rows
    {
        int c0 = lane * 8;
        f4 a0  = *(const f4*)&Ao[i * FF + c0];
        f4 a1  = *(const f4*)&Ao[i * FF + c0 + 4];
        f4 bb0 = *(const f4*)&b1s[c0];
        f4 bb1 = *(const f4*)&b1s[c0 + 4];
        for (int row = 0; row < 16; ++row) {
            int j = j0 + row;
            f4 v0 = *(const f4*)&Bo[j * FF + c0];
            f4 v1 = *(const f4*)&Bo[j * FF + c0 + 4];
            h8 o;
#pragma unroll
            for (int u = 0; u < 4; ++u) {
                o[u]     = (_Float16)lrelu(a0[u] + v0[u] + bb0[u]);
                o[4 + u] = (_Float16)lrelu(a1[u] + v1[u] + bb1[u]);
            }
            *(h8*)&z[row * PADK + c0] = o;
        }
    }

    int row_a = lane & 15;
    int khalf = lane >> 4;

    // ---- GEMM1: y(16x256) = z(16x256) @ W2^T ; 8 k-steps x 16 n-tiles
    v8f acc[16];
#pragma unroll
    for (int nt = 0; nt < 16; ++nt)
#pragma unroll
        for (int r = 0; r < 8; ++r) acc[nt][r] = 0.f;

    const _Float16* arow = &z[row_a * PADK];
#pragma unroll
    for (int ks = 0; ks < 8; ++ks) {
        v16h af = load_frag(arow, ks, khalf);
#pragma unroll
        for (int nt = 0; nt < 16; ++nt) {
            v16h bm = load_frag(&w2s[(nt * 16 + row_a) * PADK], ks, khalf);
            acc[nt] = __builtin_amdgcn_wmma_f32_16x16x32_f16(
                false, af, false, bm, (short)0, acc[nt], false, false);
        }
    }

    // epilogue: +b2, leaky, f16 -> LDS y (C/D layout: elem r -> M=r+8*hi, N=lane&15)
    int colb = lane & 15, hi = lane >> 4;
#pragma unroll
    for (int nt = 0; nt < 16; ++nt) {
        int col = nt * 16 + colb;
        float bb = b2s[col];
#pragma unroll
        for (int r = 0; r < 8; ++r)
            y[(r + 8 * hi) * PADK + col] = (_Float16)lrelu(acc[nt][r] + bb);
    }

    // ---- GEMM2: out(16x16) = y(16x256) @ Wf^T (R=16 is one n-tile)
    v8f acc2;
#pragma unroll
    for (int r = 0; r < 8; ++r) acc2[r] = 0.f;
    const _Float16* yrow = &y[row_a * PADK];
#pragma unroll
    for (int ks = 0; ks < 8; ++ks) {
        v16h af = load_frag(yrow, ks, khalf);
        v16h bm = load_frag(&wfs[row_a * PADK], ks, khalf);
        acc2 = __builtin_amdgcn_wmma_f32_16x16x32_f16(
            false, af, false, bm, (short)0, acc2, false, false);
    }
    float bfv = bfs[colb];
#pragma unroll
    for (int r = 0; r < 8; ++r) {
        int gr = i * NN + j0 + (r + 8 * hi);
        out[gr * RR + colb] = acc2[r] + bfv;
    }
}

// ---------------------------------------------------------------- host
extern "C" void kernel_launch(void* const* d_in, const int* in_sizes, int n_in,
                              void* d_out, int out_size, void* d_ws, size_t ws_size,
                              hipStream_t stream) {
    const int*   wid   = (const int*)d_in[0];
    const int*   tyid  = (const int*)d_in[1];
    const float* emb   = (const float*)d_in[2];
    const float* wty   = (const float*)d_in[3];
    const float* Wih_f = (const float*)d_in[4];
    const float* Whh_f = (const float*)d_in[5];
    const float* bih_f = (const float*)d_in[6];
    const float* bhh_f = (const float*)d_in[7];
    const float* Wih_b = (const float*)d_in[8];
    const float* Whh_b = (const float*)d_in[9];
    const float* bih_b = (const float*)d_in[10];
    const float* bhh_b = (const float*)d_in[11];
    const float* W1    = (const float*)d_in[12];
    const float* b1    = (const float*)d_in[13];
    const float* W2    = (const float*)d_in[14];
    const float* b2    = (const float*)d_in[15];
    const float* Wf    = (const float*)d_in[16];
    const float* bf    = (const float*)d_in[17];
    float* out = (float*)d_out;

    char* ws = (char*)d_ws;
    float*     x    = (float*)(ws + 0);               // 384*350*4   = 537600
    float*     gxf  = (float*)(ws + 537600);          // 384*1024*4  = 1572864
    float*     gxb  = (float*)(ws + 2110464);         // 1572864
    float*     hcat = (float*)(ws + 3683328);         // 384*512*4   = 786432
    float*     Ao   = (float*)(ws + 4469760);         // 384*256*4   = 393216
    float*     Bo   = (float*)(ws + 4862976);         // 393216
    _Float16*  w2h  = (_Float16*)(ws + 5256192);      // 256*256*2   = 131072
    _Float16*  wfh  = (_Float16*)(ws + 5387264);      // 16*256*2    = 8192
    _Float16*  whhh = (_Float16*)(ws + 5395456);      // 2*1024*256*2= 1048576  (total ~6.2 MB)

    // f16 weight prep
    k_tohalf<<<(FF * FF + 255) / 256, 256, 0, stream>>>(W2, w2h, FF * FF);
    k_tohalf<<<(RR * FF + 255) / 256, 256, 0, stream>>>(Wf, wfh, RR * FF);
    k_tohalf<<<(G4 * HH + 255) / 256, 256, 0, stream>>>(Whh_f, whhh, G4 * HH);
    k_tohalf<<<(G4 * HH + 255) / 256, 256, 0, stream>>>(Whh_b, whhh + G4 * HH, G4 * HH);

    k_embed<<<NN, 128, 0, stream>>>(wid, tyid, emb, wty, x);
    k_gx<<<NN, 1024, 0, stream>>>(x, Wih_f, bih_f, bhh_f, gxf);
    k_gx<<<NN, 1024, 0, stream>>>(x, Wih_b, bih_b, bhh_b, gxb);
    k_lstm<<<2, 1024, 0, stream>>>(gxf, gxb, whhh, hcat);
    k_ab<<<NN, 512, 0, stream>>>(hcat, W1, Ao, Bo);

    // 9216 16-row tiles, 4 waves/block -> 2304 blocks
    k_pairs<<<(NN * NN / 16) / 4, 128, 0, stream>>>(Ao, Bo, b1, w2h, b2, wfh, bf, out);
}
